// AttentionBlock_15556371546714
// MI455X (gfx1250) — matrix operand order repaired and checked by reading, not compiled, not run on hardware
//
#include <hip/hip_runtime.h>
#include <hip/hip_bf16.h>

typedef __attribute__((ext_vector_type(16))) _Float16 v16h;
typedef __attribute__((ext_vector_type(8)))  _Float16 v8h;
typedef __attribute__((ext_vector_type(8)))  float    v8f;
typedef __attribute__((ext_vector_type(4)))  int      v4i;

#define BATCH   8
#define CH      64
#define SPATIAL 4096
#define NGROUP  16
#define SOFTMAX_SCALE 0.125f   // 1/sqrt(64)

#define GLOBAL_AS __attribute__((address_space(1)))
#define LDS_AS    __attribute__((address_space(3)))

// Async Global->LDS copy path (CDNA5 async ops, ASYNCcnt).
#if defined(__gfx1250__) &&                                          \
    __has_builtin(__builtin_amdgcn_global_load_async_to_lds_b128) && \
    __has_builtin(__builtin_amdgcn_s_wait_asynccnt)
#define USE_ASYNC_LDS 1
#else
#define USE_ASYNC_LDS 0
#endif

// ---------------------------------------------------------------------------
// Fragment helpers (CDNA5 wave32 WMMA 16x16x32 f16 layouts, ISA 7.12.2)
// A 16x32 (MxK): lane m=lane%16, h=lane/16:
//   a[0..7]  = Arow[base + h*8 + 0..7]
//   a[8..15] = Arow[base + 16 + h*8 + 0..7]
// B 32x16 (KxN): lane n=lane%16, h=lane/16:
//   b[0..7]  = Bcol[base + h*8 + 0..7], b[8..15] = Bcol[base + 16 + h*8 + 0..7]
// C/D 16x16 f32: VGPR r, lane l -> element [r + 8*(l/16)][l%16]
// ---------------------------------------------------------------------------
static __device__ __forceinline__ v16h load_frag_a(const _Float16* p) {
  v8h lo = *(const v8h*)(p);
  v8h hi = *(const v8h*)(p + 16);
  v16h r;
#pragma unroll
  for (int i = 0; i < 8; ++i) { r[i] = lo[i]; r[8 + i] = hi[i]; }
  return r;
}

static __device__ __forceinline__ v16h load_frag_b(const _Float16* p) {
  v8h lo = *(const v8h*)(p);
  v8h hi = *(const v8h*)(p + 16);
  v16h r;
#pragma unroll
  for (int i = 0; i < 8; ++i) { r[i] = lo[i]; r[8 + i] = hi[i]; }
  return r;
}

static __device__ __forceinline__ v8f wmma_f16(v16h a, v16h b, v8f c) {
  // (neg_a, A, neg_b, B, c_mod, C, reuse_a, reuse_b)
  return __builtin_amdgcn_wmma_f32_16x16x32_f16(false, a, false, b, (short)0, c,
                                                false, false);
}

// ---------------------------------------------------------------------------
// Kernel 1: GroupNorm statistics. One block per (batch, group).
// ---------------------------------------------------------------------------
__global__ __launch_bounds__(256) void gn_stats(const float* __restrict__ x,
                                                float* __restrict__ mean,
                                                float* __restrict__ rstd) {
  const int bg = blockIdx.x;  // b*16 + g
  const float* base = x + (size_t)bg * 16384;
  float s = 0.f, ss = 0.f;
  for (int i = threadIdx.x; i < 16384; i += 256) {
    float v = base[i];
    s += v;
    ss += v * v;
  }
#pragma unroll
  for (int off = 16; off; off >>= 1) {
    s += __shfl_xor(s, off);
    ss += __shfl_xor(ss, off);
  }
  __shared__ float sb[8], ssb[8];
  const int wave = threadIdx.x >> 5;
  if ((threadIdx.x & 31) == 0) { sb[wave] = s; ssb[wave] = ss; }
  __syncthreads();
  if (threadIdx.x == 0) {
    float S = 0.f, SS = 0.f;
#pragma unroll
    for (int i = 0; i < 8; ++i) { S += sb[i]; SS += ssb[i]; }
    float m = S * (1.f / 16384.f);
    float var = SS * (1.f / 16384.f) - m * m;
    mean[bg] = m;
    rstd[bg] = rsqrtf(var + 1e-5f);
  }
}

// ---------------------------------------------------------------------------
// Kernel 2: normalize + QKV projection. Q,K token-major [b][n][c] f16;
// V channel-major [b][c][n] f16 so all WMMA fragment loads are contiguous.
// ---------------------------------------------------------------------------
__global__ __launch_bounds__(256) void qkv_kernel(
    const float* __restrict__ x, const float* __restrict__ mean,
    const float* __restrict__ rstd, const float* __restrict__ nw,
    const float* __restrict__ nb, const float* __restrict__ qw,
    const float* __restrict__ qb, _Float16* __restrict__ Q,
    _Float16* __restrict__ K, _Float16* __restrict__ V) {
  __shared__ float xn[64 * 64];  // [c][p]
  const int blk = blockIdx.x;
  const int b = blk >> 6;
  const int p0 = (blk & 63) << 6;
  const int tid = threadIdx.x;

#pragma unroll
  for (int i = 0; i < 16; ++i) {
    int f = tid + i * 256;
    int c = f >> 6, p = f & 63;
    int bg = b * NGROUP + (c >> 2);
    float v = x[((size_t)(b * CH + c)) * SPATIAL + p0 + p];
    xn[f] = (v - mean[bg]) * rstd[bg] * nw[c] + nb[c];
  }
  __syncthreads();

  const int p = tid & 63;
  const int dbase = tid >> 6;  // uniform per wave -> scalar weight loads
#pragma unroll 4
  for (int i = 0; i < 48; ++i) {
    int d = dbase + (i << 2);
    const float* wr = qw + d * CH;
    float acc = qb[d];
#pragma unroll
    for (int c = 0; c < CH; ++c) acc += xn[(c << 6) + p] * wr[c];
    _Float16 hv = (_Float16)acc;
    size_t n = (size_t)b * SPATIAL + p0 + p;
    if (d < 64)
      Q[n * CH + d] = hv;
    else if (d < 128)
      K[n * CH + (d - 64)] = hv;
    else
      V[((size_t)b * CH + (d - 128)) * SPATIAL + p0 + p] = hv;
  }
}

// ---------------------------------------------------------------------------
// K/V tile staging into LDS: async Global->LDS (ASYNCcnt) or sync fallback.
// 4 x b128 per thread per tile (K: 2, V: 2).
// ---------------------------------------------------------------------------
static __device__ __forceinline__ void stage_tile(
    const _Float16* __restrict__ Kg, const _Float16* __restrict__ Vg, int b,
    int kb, _Float16* kdst, _Float16* vdst, int tid) {
  const uint4* ks = (const uint4*)(Kg + ((size_t)b * SPATIAL + kb * 64) * CH);
  uint4* kd = (uint4*)kdst;
  uint4* vd = (uint4*)vdst;
#if USE_ASYNC_LDS
#pragma unroll
  for (int i = 0; i < 2; ++i) {
    int idx = tid + i * 256;
    __builtin_amdgcn_global_load_async_to_lds_b128(
        (GLOBAL_AS v4i*)(ks + idx), (LDS_AS v4i*)(kd + idx), 0, 0);
  }
#pragma unroll
  for (int i = 0; i < 2; ++i) {
    int f = tid + i * 256;
    int row = f >> 3, chk = f & 7;
    const uint4* vs =
        (const uint4*)(Vg + ((size_t)b * CH + row) * SPATIAL + kb * 64 + chk * 8);
    __builtin_amdgcn_global_load_async_to_lds_b128(
        (GLOBAL_AS v4i*)vs, (LDS_AS v4i*)(vd + f), 0, 0);
  }
#else
  kd[tid] = ks[tid];
  kd[tid + 256] = ks[tid + 256];
#pragma unroll
  for (int i = 0; i < 2; ++i) {
    int f = tid + i * 256;
    int row = f >> 3, chk = f & 7;
    vd[f] = *(const uint4*)(Vg + ((size_t)b * CH + row) * SPATIAL + kb * 64 +
                            chk * 8);
  }
  if (kb + 1 < 64 && tid < 64)  // prefetch next K tile (global_prefetch_b8)
    __builtin_prefetch(Kg + ((size_t)b * SPATIAL + (kb + 1) * 64) * CH + tid * 64,
                       0, 0);
#endif
}

static __device__ __forceinline__ void wait_async_tail() {
#if USE_ASYNC_LDS
  __builtin_amdgcn_s_wait_asynccnt(4);  // older tile's 4 loads complete
#endif
}
static __device__ __forceinline__ void wait_async_all() {
#if USE_ASYNC_LDS
  __builtin_amdgcn_s_wait_asynccnt(0);
#endif
}

// ---------------------------------------------------------------------------
// Kernel 3: flash attention. Grid (32 query-blocks, 8 batches), 256 threads,
// 8 waves; wave w owns query rows [qbase+16w, qbase+16w+16).
// Double-buffered K/V tiles; per 64-key iteration: 8 WMMAs for S, online
// softmax in C-layout, P via per-wave LDS, 8 WMMAs for O.
// ---------------------------------------------------------------------------
__global__ __launch_bounds__(256) void attn_kernel(
    const _Float16* __restrict__ Qg, const _Float16* __restrict__ Kg,
    const _Float16* __restrict__ Vg, _Float16* __restrict__ Og) {
  __shared__ alignas(16) _Float16 Kt[2][64 * 64];  // [buf][key][c]
  __shared__ alignas(16) _Float16 Vt[2][64 * 64];  // [buf][c][key]
  __shared__ alignas(16) _Float16 Pl[8][16 * 64];  // per-wave P strip

  const int b = blockIdx.y;
  const int qbase = blockIdx.x * 128;
  const int tid = threadIdx.x;
  const int wave = tid >> 5;
  const int lane = tid & 31;
  const int h = lane >> 4;
  const int n16 = lane & 15;

  // Q fragments for this wave's 16 query rows (K-chunks of 32 channels).
  const _Float16* qrow =
      Qg + ((size_t)b * SPATIAL + qbase + wave * 16 + n16) * CH;
  const v16h qf0 = load_frag_a(qrow + 0 * 32 + h * 8);
  const v16h qf1 = load_frag_a(qrow + 1 * 32 + h * 8);

  float m_state[8], l_state[8];
#pragma unroll
  for (int r = 0; r < 8; ++r) { m_state[r] = -1e30f; l_state[r] = 0.f; }
  v8f oacc[4] = {};

  stage_tile(Kg, Vg, b, 0, Kt[0], Vt[0], tid);  // pipeline prologue

  for (int kb = 0; kb < 64; ++kb) {
    const int cur = kb & 1;
    if (kb + 1 < 64) {
      stage_tile(Kg, Vg, b, kb + 1, Kt[cur ^ 1], Vt[cur ^ 1], tid);
      wait_async_tail();   // tile kb complete (in-order async completion)
    } else {
      wait_async_all();
    }
    __syncthreads();       // publish tile kb across waves

    const _Float16* KtC = Kt[cur];
    const _Float16* VtC = Vt[cur];

    // S = Q * K^T  (4 key tiles of 16, K-dim 64 = 2 WMMA chunks)
    v8f sacc[4] = {};
#pragma unroll
    for (int t = 0; t < 4; ++t) {
      const _Float16* kp = KtC + (t * 16 + n16) * CH;
      sacc[t] = wmma_f16(qf0, load_frag_b(kp + 0 * 32 + h * 8), sacc[t]);
      sacc[t] = wmma_f16(qf1, load_frag_b(kp + 1 * 32 + h * 8), sacc[t]);
    }

    // Online softmax on C-layout accumulators (row = r + 8h, lane group h).
    float mnew[8], corr[8], rsum[8];
#pragma unroll
    for (int r = 0; r < 8; ++r) {
      float mx = fmaxf(fmaxf(sacc[0][r], sacc[1][r]),
                       fmaxf(sacc[2][r], sacc[3][r])) * SOFTMAX_SCALE;
#pragma unroll
      for (int off = 1; off < 16; off <<= 1) mx = fmaxf(mx, __shfl_xor(mx, off));
      mnew[r] = fmaxf(m_state[r], mx);
      corr[r] = __expf(m_state[r] - mnew[r]);
      m_state[r] = mnew[r];
      rsum[r] = 0.f;
    }

    _Float16* pstrip = Pl[wave];
#pragma unroll
    for (int t = 0; t < 4; ++t) {
#pragma unroll
      for (int r = 0; r < 8; ++r) {
        float pv = __expf(sacc[t][r] * SOFTMAX_SCALE - mnew[r]);
        rsum[r] += pv;
        pstrip[(r + 8 * h) * CH + t * 16 + n16] = (_Float16)pv;
      }
    }
#pragma unroll
    for (int r = 0; r < 8; ++r) {
      float rs = rsum[r];
#pragma unroll
      for (int off = 1; off < 16; off <<= 1) rs += __shfl_xor(rs, off);
      l_state[r] = l_state[r] * corr[r] + rs;
#pragma unroll
      for (int t = 0; t < 4; ++t) oacc[t][r] *= corr[r];
    }

    // O += P * V   (P from per-wave LDS as A-frags; V^T tile -> contig B)
    const _Float16* prow = pstrip + n16 * CH;
#pragma unroll
    for (int ct = 0; ct < 4; ++ct) {
      const _Float16* vp = VtC + (ct * 16 + n16) * CH;
      oacc[ct] = wmma_f16(load_frag_a(prow + 0 * 32 + h * 8),
                          load_frag_b(vp + 0 * 32 + h * 8), oacc[ct]);
      oacc[ct] = wmma_f16(load_frag_a(prow + 1 * 32 + h * 8),
                          load_frag_b(vp + 1 * 32 + h * 8), oacc[ct]);
    }
    __syncthreads();  // all waves done reading tile kb before it is rewritten
  }

  // Epilogue: normalize rows and store att output [b][n][c] f16.
#pragma unroll
  for (int t = 0; t < 4; ++t) {
#pragma unroll
    for (int r = 0; r < 8; ++r) {
      float ov = oacc[t][r] / l_state[r];
      int row = qbase + wave * 16 + r + 8 * h;
      Og[((size_t)b * SPATIAL + row) * CH + t * 16 + n16] = (_Float16)ov;
    }
  }
}

// ---------------------------------------------------------------------------
// Kernel 4: output projection + bias + residual.
// ---------------------------------------------------------------------------
__global__ __launch_bounds__(256) void proj_kernel(
    const float* __restrict__ x, const _Float16* __restrict__ att,
    const float* __restrict__ pw, const float* __restrict__ pb,
    float* __restrict__ out) {
  int idx = blockIdx.x * 256 + threadIdx.x;  // b*262144 + d*4096 + p
  int p = idx & 4095;
  int d = (idx >> 12) & 63;
  int b = idx >> 18;
  const _Float16* ar = att + ((size_t)b * SPATIAL + p) * CH;
  const float* wr = pw + d * CH;  // d uniform per wave -> scalar loads
  float acc = pb[d];
#pragma unroll
  for (int c = 0; c < CH; ++c) acc += (float)ar[c] * wr[c];
  out[idx] = x[idx] + acc;
}

// ---------------------------------------------------------------------------
extern "C" void kernel_launch(void* const* d_in, const int* in_sizes, int n_in,
                              void* d_out, int out_size, void* d_ws,
                              size_t ws_size, hipStream_t stream) {
  (void)in_sizes; (void)n_in; (void)out_size; (void)ws_size;
  const float* x      = (const float*)d_in[0];
  const float* norm_w = (const float*)d_in[1];
  const float* norm_b = (const float*)d_in[2];
  const float* qkv_w  = (const float*)d_in[3];
  const float* qkv_b  = (const float*)d_in[4];
  const float* proj_w = (const float*)d_in[5];
  const float* proj_b = (const float*)d_in[6];
  float* out = (float*)d_out;

  // Workspace layout (16.8 MB): stats | Q | K | V | att.
  char* ws = (char*)d_ws;
  float* meanb = (float*)ws;   // 128 f32
  float* rstdb = meanb + 128;  // 128 f32
  const size_t qkv_elems = (size_t)BATCH * SPATIAL * CH;
  _Float16* Q = (_Float16*)(ws + 4096);
  _Float16* K = Q + qkv_elems;
  _Float16* V = K + qkv_elems;
  _Float16* A = V + qkv_elems;

  gn_stats<<<BATCH * NGROUP, 256, 0, stream>>>(x, meanb, rstdb);
  qkv_kernel<<<BATCH * (SPATIAL / 64), 256, 0, stream>>>(
      x, meanb, rstdb, norm_w, norm_b, qkv_w, qkv_b, Q, K, V);
  attn_kernel<<<dim3(SPATIAL / 128, BATCH), 256, 0, stream>>>(Q, K, V, A);
  proj_kernel<<<(BATCH * CH * SPATIAL) / 256, 256, 0, stream>>>(x, A, proj_w,
                                                                proj_b, out);
}
